// ODEBlock_20736102105354
// MI455X (gfx1250) — matrix operand order repaired and checked by reading, not compiled, not run on hardware
//
#include <hip/hip_runtime.h>
#include <math.h>

// ---------------------------------------------------------------------------
// ODEBlock (time-conditioned transformer block) for gfx1250 (MI455X).
// All matmuls run on v_wmma_f32_16x16x32_bf16 (fp32 accumulate); LayerNorm /
// softmax statistics stay fp32.  GEMM stages A/B tiles into LDS with the
// CDNA5 async engine (global_load_async_to_lds_b128 + ASYNCcnt), double
// buffered with one workgroup barrier per K-step.  Attention is flash-style,
// computed transposed (S^T = K * Q^T) so the softmax reduction is in-lane
// plus one shfl_xor(16).
// ---------------------------------------------------------------------------

typedef __attribute__((ext_vector_type(16))) __bf16        v16bf;
typedef __attribute__((ext_vector_type(8)))  float         v8f;
typedef __attribute__((ext_vector_type(4)))  unsigned int  u32x4;

union Frag32B {
  v16bf         v;
  u32x4         q[2];
  unsigned int  d[8];
};

__device__ __forceinline__ unsigned short f2bf(float f) {
  unsigned int u = __builtin_bit_cast(unsigned int, f);
  u += 0x7FFFu + ((u >> 16) & 1u);               // round-to-nearest-even
  return (unsigned short)(u >> 16);
}

__device__ __forceinline__ float gelu_exact(float x) {
  return 0.5f * x * (1.0f + erff(x * 0.70710678118654752f));
}

// CDNA5 async global->LDS copy, 16B per lane.  lds_off = LDS byte offset
// (low 32 bits of the generic pointer to a __shared__ object).
__device__ __forceinline__ void async_ld_b128(unsigned lds_off, const void* gaddr) {
  asm volatile("global_load_async_to_lds_b128 %0, %1, off"
               :: "v"(lds_off), "v"((unsigned long long)(size_t)gaddr)
               : "memory");
}

__device__ __forceinline__ void wait_asynccnt0() {
#if __has_builtin(__builtin_amdgcn_s_wait_asynccnt)
  __builtin_amdgcn_s_wait_asynccnt(0);
#else
  asm volatile("s_wait_asynccnt 0" ::: "memory");
#endif
}

// ------------------------- time MLP: t -> t_emb[B][64] ----------------------
__global__ void k_time_mlp(const float* __restrict__ t,  const float* __restrict__ w1,
                           const float* __restrict__ b1, const float* __restrict__ w2,
                           const float* __restrict__ b2, float* __restrict__ temb) {
  __shared__ float h1[2][64];
  int tid = threadIdx.x;                // 0..127
  int b = tid >> 6, i = tid & 63;
  h1[b][i] = gelu_exact(t[b] * w1[i] + b1[i]);
  __syncthreads();
  float acc = b2[i];
  for (int j = 0; j < 64; ++j) acc += h1[b][j] * w2[j * 64 + i];
  temb[b * 64 + i] = acc;
}

// ------------- weight convert+transpose: w[K][N] f32 -> wt[N][K] bf16 -------
__global__ void k_w_to_bf16t(const float* __restrict__ w, unsigned short* __restrict__ wt,
                             int K, int N) {
  size_t idx = (size_t)blockIdx.x * 256 + threadIdx.x;   // over K*N, read-coalesced
  int kk = (int)(idx / (size_t)N);
  int nn = (int)(idx % (size_t)N);
  wt[(size_t)nn * K + kk] = f2bf(w[idx]);
}

// ---- fused concat(x[row][1024], temb[b][64]) + LayerNorm -> bf16[row][1088]
__global__ __launch_bounds__(256) void k_ln_concat(
    const float* __restrict__ X, const float* __restrict__ temb,
    const float* __restrict__ w, const float* __restrict__ bias,
    unsigned short* __restrict__ out) {
  const int C = 1088;
  int row = blockIdx.x, tid = threadIdx.x;
  int b = row >> 11;                               // T = 2048
  float v[5];
  float s = 0.f, s2 = 0.f;
  int nj = 0;
  for (int i = 0; i < 5; ++i) {
    int j = tid + 256 * i;
    if (j < C) {
      float x = (j < 1024) ? X[(size_t)row * 1024 + j] : temb[b * 64 + (j - 1024)];
      v[i] = x; s += x; s2 += x * x; nj = i + 1;
    }
  }
  for (int off = 16; off; off >>= 1) { s += __shfl_xor(s, off, 32); s2 += __shfl_xor(s2, off, 32); }
  __shared__ float rs[8], rq[8], stat[2];
  int wid = tid >> 5, lane = tid & 31;
  if (lane == 0) { rs[wid] = s; rq[wid] = s2; }
  __syncthreads();
  if (tid == 0) {
    float a = 0.f, c = 0.f;
    for (int wdx = 0; wdx < 8; ++wdx) { a += rs[wdx]; c += rq[wdx]; }
    float mean = a / C;
    stat[0] = mean;
    stat[1] = rsqrtf(c / C - mean * mean + 1e-5f);
  }
  __syncthreads();
  float mean = stat[0], rstd = stat[1];
  for (int i = 0; i < nj; ++i) {
    int j = tid + 256 * i;
    out[(size_t)row * C + j] = f2bf((v[i] - mean) * rstd * w[j] + bias[j]);
  }
}

// --------------------------- generic bf16 WMMA GEMM -------------------------
// C[M][N] = A[M][K] * Bt[N][K]^T + bias, optional exact GELU, bf16 or f32 out.
// Block: 256 thr = 8 waves (2M x 4N), wave tile 32x32, block tile 64x128,
// K step 32.  LDS double-buffered, filled by async global->LDS (ASYNCcnt),
// one barrier per K-step.
__global__ __launch_bounds__(256) void k_gemm_bf16(
    const unsigned short* __restrict__ A,    // [M][K] bf16 row-major
    const unsigned short* __restrict__ Bt,   // [N][K] bf16 row-major (transposed W)
    const float* __restrict__ bias,          // [N]
    void* __restrict__ Cout, int M, int N, int K, int do_gelu, int out_bf16) {
  __shared__ unsigned short As[2][64 * 40];  // 32 k-cols + pad (16B-aligned rows)
  __shared__ unsigned short Bs[2][128 * 40];
  const int tid = threadIdx.x;
  const int wid = tid >> 5, lane = tid & 31;
  const int wm = wid >> 2, wn = wid & 3;
  const int hl = lane >> 4, lr = lane & 15;
  const int rowBase = blockIdx.y * 64;
  const int colBase = blockIdx.x * 128;

  v8f acc[2][2] = {};

  const int sr = tid >> 2;                   // staging row 0..63
  const int sc = (tid & 3) * 8;              // staging col 0,8,16,24
  const unsigned short* gA  = A  + (size_t)(rowBase + sr) * K + sc;
  const unsigned short* gB0 = Bt + (size_t)(colBase + sr) * K + sc;
  const unsigned short* gB1 = Bt + (size_t)(colBase + 64 + sr) * K + sc;

  unsigned ldsA[2], ldsB0[2], ldsB1[2];
  for (int bx = 0; bx < 2; ++bx) {
    ldsA[bx]  = (unsigned)(size_t)&As[bx][sr * 40 + sc];
    ldsB0[bx] = (unsigned)(size_t)&Bs[bx][sr * 40 + sc];
    ldsB1[bx] = (unsigned)(size_t)&Bs[bx][(sr + 64) * 40 + sc];
  }

  // stage tile (3 x 16B per thread) into buffer bx via async engine
  auto stage = [&](int k0, int bx) {
    async_ld_b128(ldsA[bx],  gA  + k0);
    async_ld_b128(ldsB0[bx], gB0 + k0);
    async_ld_b128(ldsB1[bx], gB1 + k0);
  };

  const int nk = K >> 5;
  stage(0, 0);
  for (int it = 0; it < nk; ++it) {
    wait_asynccnt0();                        // my buf[it&1] writes landed in LDS
    __syncthreads();                         // everyone's landed; prev reads done
    if (it + 1 < nk) stage((it + 1) << 5, (it + 1) & 1);
    const int bx = it & 1;

    Frag32B af[2], bfr[2];
    for (int am = 0; am < 2; ++am) {         // A frag: lanes<16 K{0..7,16..23}
      int r = wm * 32 + am * 16 + lr;
      af[am].q[0] = *(const u32x4*)&As[bx][r * 40 + hl * 8];
      af[am].q[1] = *(const u32x4*)&As[bx][r * 40 + 16 + hl * 8];
    }
    for (int an = 0; an < 2; ++an) {         // B frag: lanes<16 K{0..7,16..23}
      int c = wn * 32 + an * 16 + lr;
      bfr[an].q[0] = *(const u32x4*)&Bs[bx][c * 40 + hl * 8];
      bfr[an].q[1] = *(const u32x4*)&Bs[bx][c * 40 + 16 + hl * 8];
    }
    for (int am = 0; am < 2; ++am)
      for (int an = 0; an < 2; ++an)
        acc[am][an] = __builtin_amdgcn_wmma_f32_16x16x32_bf16(
            false, af[am].v, false, bfr[an].v, (short)0, acc[am][an], false, false);
  }

  for (int am = 0; am < 2; ++am) {
    for (int an = 0; an < 2; ++an) {
      int n = colBase + wn * 32 + an * 16 + lr;
      float bv = bias[n];
      for (int r = 0; r < 8; ++r) {
        int m = rowBase + wm * 32 + am * 16 + r + 8 * hl;
        float x = acc[am][an][r] + bv;
        if (do_gelu) x = gelu_exact(x);
        if (out_bf16) ((unsigned short*)Cout)[(size_t)m * N + n] = f2bf(x);
        else          ((float*)Cout)[(size_t)m * N + n] = x;
      }
    }
  }
}

// ------ repack qkv[4096][3072] bf16 -> q,k:[B,H,T,64], vt:[B,H,64,T] --------
__global__ void k_repack_qkv(const unsigned short* __restrict__ qkv,
                             unsigned short* __restrict__ q,
                             unsigned short* __restrict__ k,
                             unsigned short* __restrict__ vt) {
  size_t idx = (size_t)blockIdx.x * 256 + threadIdx.x;   // < 4096*3072
  int row = (int)(idx / 3072);
  int c   = (int)(idx % 3072);
  int b = row >> 11, tt = row & 2047;
  unsigned short val = qkv[idx];
  if (c < 1024) {
    int h = c >> 6, d = c & 63;
    q[(((size_t)(b * 16 + h) * 2048 + tt) << 6) + d] = val;
  } else if (c < 2048) {
    int cc = c - 1024, h = cc >> 6, d = cc & 63;
    k[(((size_t)(b * 16 + h) * 2048 + tt) << 6) + d] = val;
  } else {
    int cc = c - 2048, h = cc >> 6, d = cc & 63;
    vt[((size_t)(b * 16 + h) * 64 + d) * 2048 + tt] = val;
  }
}

// ------------------ causal flash attention, one wave per 16-query tile ------
// S^T(32k x 16q) = K*Q^T via WMMA; softmax over keys is in-lane + shfl_xor(16);
// O^T(64d x 16q) += V^T * P^T via WMMA.  hd = 64, scale = 0.125.
__global__ __launch_bounds__(256) void k_attention(
    const unsigned short* __restrict__ q, const unsigned short* __restrict__ k,
    const unsigned short* __restrict__ vt, unsigned short* __restrict__ y) {
  const int tid = threadIdx.x;
  const int wid = tid >> 5, lane = tid & 31;
  const int hl = lane >> 4, lr = lane & 15;
  int task = blockIdx.x * 8 + wid;               // < 4096 = B*H*(T/16)
  int jq = task & 127;
  int h  = (task >> 7) & 15;
  int b  = task >> 11;
  const int qt = jq * 16;
  const unsigned short* qbh = q  + ((size_t)(b * 16 + h) * 2048) * 64;
  const unsigned short* kbh = k  + ((size_t)(b * 16 + h) * 2048) * 64;
  const unsigned short* vbh = vt + ((size_t)(b * 16 + h) * 64) * 2048;

  Frag32B qf[2];                                 // Q^T as B-frags, dims 0..31 / 32..63
  for (int kk = 0; kk < 2; ++kk) {
    const unsigned short* p = qbh + (size_t)(qt + lr) * 64 + kk * 32 + hl * 8;
    qf[kk].q[0] = *(const u32x4*)p;
    qf[kk].q[1] = *(const u32x4*)(p + 16);
  }

  v8f o[4] = {};
  float mi = -3.0e38f, li = 0.f;
  const int qcur = qt + lr;
  const int kbEnd = ((qt + 15) >> 5) << 5;

  for (int kb = 0; kb <= kbEnd; kb += 32) {
    if (kb + 32 <= kbEnd) {                      // prefetch next K/V block
      __builtin_prefetch((const void*)(kbh + (size_t)(kb + 32) * 64 + lane * 64), 0, 2);
      __builtin_prefetch((const void*)(vbh + (size_t)lane * 2048 + kb + 32), 0, 2);
    }
    v8f s0 = {}, s1 = {};
    for (int kk = 0; kk < 2; ++kk) {
      Frag32B ka;
      const unsigned short* p0 = kbh + (size_t)(kb + lr) * 64 + kk * 32 + hl * 8;
      ka.q[0] = *(const u32x4*)p0;
      ka.q[1] = *(const u32x4*)(p0 + 16);
      s0 = __builtin_amdgcn_wmma_f32_16x16x32_bf16(false, ka.v, false, qf[kk].v,
                                                   (short)0, s0, false, false);
      const unsigned short* p1 = kbh + (size_t)(kb + 16 + lr) * 64 + kk * 32 + hl * 8;
      ka.q[0] = *(const u32x4*)p1;
      ka.q[1] = *(const u32x4*)(p1 + 16);
      s1 = __builtin_amdgcn_wmma_f32_16x16x32_bf16(false, ka.v, false, qf[kk].v,
                                                   (short)0, s1, false, false);
    }
    float sv0[8], sv1[8];
    float mx = -3.0e38f;
    for (int r = 0; r < 8; ++r) {
      int key0 = kb + r + 8 * hl;                // S^T rows are keys
      float a0 = s0[r] * 0.125f;
      float a1 = s1[r] * 0.125f;
      a0 = (key0      <= qcur) ? a0 : -3.0e38f;
      a1 = (key0 + 16 <= qcur) ? a1 : -3.0e38f;
      sv0[r] = a0; sv1[r] = a1;
      mx = fmaxf(mx, fmaxf(a0, a1));
    }
    mx = fmaxf(mx, __shfl_xor(mx, 16, 32));
    float mnew = fmaxf(mi, mx);
    float corr = __expf(mi - mnew);
    float lsum = 0.f;
    unsigned int pk0[4], pk1[4];
    for (int i = 0; i < 4; ++i) {
      float p0a = __expf(sv0[2 * i]     - mnew);
      float p0b = __expf(sv0[2 * i + 1] - mnew);
      float p1a = __expf(sv1[2 * i]     - mnew);
      float p1b = __expf(sv1[2 * i + 1] - mnew);
      lsum += p0a + p0b + p1a + p1b;
      pk0[i] = (unsigned int)f2bf(p0a) | ((unsigned int)f2bf(p0b) << 16);
      pk1[i] = (unsigned int)f2bf(p1a) | ((unsigned int)f2bf(p1b) << 16);
    }
    lsum += __shfl_xor(lsum, 16, 32);
    li = li * corr + lsum;
    mi = mnew;
    Frag32B pf;                                  // P^T as B-frag (32 keys x 16 queries)
    for (int i = 0; i < 4; ++i) {
      unsigned int o0 = (unsigned int)__shfl_xor((int)pk0[i], 16, 32);
      unsigned int o1 = (unsigned int)__shfl_xor((int)pk1[i], 16, 32);
      pf.d[i]     = hl ? o1     : pk0[i];
      pf.d[4 + i] = hl ? pk1[i] : o0;
    }
    for (int dt = 0; dt < 4; ++dt) {
      o[dt] = o[dt] * corr;
      Frag32B va;                                // V^T frag (16 dims x 32 keys)
      const unsigned short* vp = vbh + (size_t)(dt * 16 + lr) * 2048 + kb + hl * 16;
      va.q[0] = *(const u32x4*)vp;
      va.q[1] = *(const u32x4*)(vp + 8);
      o[dt] = __builtin_amdgcn_wmma_f32_16x16x32_bf16(false, va.v, false, pf.v,
                                                      (short)0, o[dt], false, false);
    }
  }

  float inv = 1.0f / li;
  unsigned short* yp = y + (size_t)(b * 2048 + qt + lr) * 1024 + h * 64;
  for (int dt = 0; dt < 4; ++dt) {
    unsigned int* w32 = (unsigned int*)(yp + dt * 16 + 8 * hl);
    for (int i = 0; i < 4; ++i) {
      unsigned int lo = f2bf(o[dt][2 * i]     * inv);
      unsigned int hi = f2bf(o[dt][2 * i + 1] * inv);
      w32[i] = lo | (hi << 16);
    }
  }
}

// ---------------------------------------------------------------------------
extern "C" void kernel_launch(void* const* d_in, const int* in_sizes, int n_in,
                              void* d_out, int out_size, void* d_ws, size_t ws_size,
                              hipStream_t stream) {
  (void)in_sizes; (void)n_in; (void)out_size; (void)ws_size;
  const float* z      = (const float*)d_in[0];
  const float* t      = (const float*)d_in[1];
  const float* w_t1   = (const float*)d_in[2];
  const float* b_t1   = (const float*)d_in[3];
  const float* w_t2   = (const float*)d_in[4];
  const float* b_t2   = (const float*)d_in[5];
  const float* ln_a_w = (const float*)d_in[6];
  const float* ln_a_b = (const float*)d_in[7];
  const float* w_attn = (const float*)d_in[8];
  const float* b_attn = (const float*)d_in[9];
  const float* w_proj = (const float*)d_in[10];
  const float* b_proj = (const float*)d_in[11];
  const float* ln_m_w = (const float*)d_in[12];
  const float* ln_m_b = (const float*)d_in[13];
  const float* w_fc   = (const float*)d_in[14];
  const float* b_fc   = (const float*)d_in[15];
  const float* w_fc2  = (const float*)d_in[16];
  const float* b_fc2  = (const float*)d_in[17];
  float* out = (float*)d_out;

  // workspace arena (~91 MB) with phase reuse
  char* ws = (char*)d_ws;
  const size_t OFF_XLN = 4096;                         // x_ln / m_ln   (8,912,896)
  const size_t OFF_WT1 = OFF_XLN + 8912896;            // wT_attn / wT_fc (8,912,896)
  const size_t OFF_BIG = OFF_WT1 + 8912896;            // qkv+q / act   (33,554,432)
  const size_t OFF_K   = OFF_BIG + 33554432;           // k / wT_fc2    (8,388,608)
  const size_t OFF_VT  = OFF_K   + 8388608;            // vt            (8,388,608)
  const size_t OFF_Y   = OFF_VT  + 8388608;            // y bf16        (8,388,608)
  const size_t OFF_WTP = OFF_Y   + 8388608;            // wT_proj       (2,097,152)
  const size_t OFF_H   = OFF_WTP + 2097152;            // h f32        (16,777,216)

  float*          temb  = (float*)(ws);
  unsigned short* xln   = (unsigned short*)(ws + OFF_XLN);
  unsigned short* wt1   = (unsigned short*)(ws + OFF_WT1);
  unsigned short* qkv   = (unsigned short*)(ws + OFF_BIG);
  unsigned short* qbuf  = (unsigned short*)(ws + OFF_BIG + 25165824);
  unsigned short* kbuf  = (unsigned short*)(ws + OFF_K);
  unsigned short* vtbuf = (unsigned short*)(ws + OFF_VT);
  unsigned short* ybuf  = (unsigned short*)(ws + OFF_Y);
  unsigned short* wtp   = (unsigned short*)(ws + OFF_WTP);
  float*          hbuf  = (float*)(ws + OFF_H);
  unsigned short* mln   = xln;    // reuse after qkv GEMM
  unsigned short* wtfc  = wt1;    // reuse after qkv GEMM
  unsigned short* act   = qkv;    // reuse (spans qkv+q) after attention
  unsigned short* wtfc2 = kbuf;   // reuse after attention

  k_time_mlp<<<1, 128, 0, stream>>>(t, w_t1, b_t1, w_t2, b_t2, temb);
  k_w_to_bf16t<<<(1088 * 3072) / 256, 256, 0, stream>>>(w_attn, wt1, 1088, 3072);
  k_ln_concat<<<4096, 256, 0, stream>>>(z, temb, ln_a_w, ln_a_b, xln);
  k_gemm_bf16<<<dim3(3072 / 128, 4096 / 64), 256, 0, stream>>>(
      xln, wt1, b_attn, qkv, 4096, 3072, 1088, 0, 1);
  k_repack_qkv<<<(4096 * 3072) / 256, 256, 0, stream>>>(qkv, qbuf, kbuf, vtbuf);
  k_attention<<<512, 256, 0, stream>>>(qbuf, kbuf, vtbuf, ybuf);
  k_w_to_bf16t<<<(1024 * 1024) / 256, 256, 0, stream>>>(w_proj, wtp, 1024, 1024);
  k_gemm_bf16<<<dim3(1024 / 128, 4096 / 64), 256, 0, stream>>>(
      ybuf, wtp, b_proj, hbuf, 4096, 1024, 1024, 0, 0);
  k_ln_concat<<<4096, 256, 0, stream>>>(hbuf, temb, ln_m_w, ln_m_b, mln);
  k_w_to_bf16t<<<(1088 * 4096) / 256, 256, 0, stream>>>(w_fc, wtfc, 1088, 4096);
  k_gemm_bf16<<<dim3(4096 / 128, 4096 / 64), 256, 0, stream>>>(
      mln, wtfc, b_fc, act, 4096, 4096, 1088, 1, 1);
  k_w_to_bf16t<<<(4096 * 1024) / 256, 256, 0, stream>>>(w_fc2, wtfc2, 4096, 1024);
  k_gemm_bf16<<<dim3(1024 / 128, 4096 / 64), 256, 0, stream>>>(
      act, wtfc2, b_fc2, out, 4096, 1024, 4096, 0, 0);
}
